// AceStepAttention_47605417509333
// MI455X (gfx1250) — compile-verified, matrix operand order checked by reading, not run on hardware
//
#include <hip/hip_runtime.h>
#include <hip/hip_bf16.h>

#define B_   4
#define S_   2048
#define HID_ 2048
#define H_   16
#define KV_  4
#define D_   128
#define G_   (H_ / KV_)
#define SCALE_ 0.08838834764831845f
#define EPS_ 1e-6f

typedef __attribute__((ext_vector_type(16))) __bf16 v16bf;
typedef __attribute__((ext_vector_type(8)))  __bf16 v8bf;
typedef __attribute__((ext_vector_type(8)))  float  v8f;

#if defined(__HIP_DEVICE_COMPILE__) && __has_builtin(__builtin_amdgcn_tensor_load_to_lds)
#define USE_TDM 1
#else
#define USE_TDM 0
#endif

// Load one 16-bf16 WMMA operand fragment from a 32-element contiguous K-chunk.
// Per ISA 16-bit A/B layout: lanes 0-15 hold K{0..7,16..23}, lanes 16-31 hold K{8..15,24..31}.
__device__ __forceinline__ v16bf ld_frag(const __bf16* p, int lane) {
    const __bf16* q = p + ((lane & 16) ? 8 : 0);
    v8bf lo = *(const v8bf*)(q);
    v8bf hi = *(const v8bf*)(q + 16);
    return __builtin_shufflevector(lo, hi, 0,1,2,3,4,5,6,7,8,9,10,11,12,13,14,15);
}

__device__ __forceinline__ v8f wmma_bf16(v16bf a, v16bf b, v8f c) {
    return __builtin_amdgcn_wmma_f32_16x16x32_bf16(false, a, false, b, (short)0, c, false, false);
}

#if USE_TDM
typedef __attribute__((ext_vector_type(4))) unsigned int u32x4_t;
typedef __attribute__((ext_vector_type(8))) int i32x8_t;
typedef __attribute__((ext_vector_type(4))) int i32x4_t;

// TDM 2D tile load: rows x 32 bf16 elements, row stride = row_stride_elems, into LDS.
// Builds D# group0/group1 per CDNA5 ISA 8.3/8.4; groups 2/3 (and the extra descriptor
// words of the 6-arg builtin form) zero for <=2D tensors.
__device__ __forceinline__ void tdm_load_tile(const __bf16* gsrc, __bf16* ldst,
                                              int rows, int row_stride_elems) {
    unsigned long long ga = (unsigned long long)(size_t)gsrc;
    unsigned lo = (unsigned)(size_t)ldst;           // generic addr low 32 bits = LDS byte offset
    u32x4_t g0;
    g0[0] = 1u;                                     // count=1, user descriptor
    g0[1] = lo;                                     // lds_addr
    g0[2] = (unsigned)ga;                           // global_addr[31:0]
    g0[3] = ((unsigned)(ga >> 32) & 0x01ffffffu) | 0x80000000u;  // global_addr[56:32] | type=2
    unsigned td0 = (unsigned)row_stride_elems;      // tensor_dim0 (width, elems)
    unsigned td1 = (unsigned)rows;                  // tensor_dim1
    unsigned long long st0 = (unsigned long long)row_stride_elems;
    i32x8_t g1;
    g1[0] = (int)(1u << 16);                        // workgroup_mask=0, data_size=1 (2 bytes)
    g1[1] = (int)((td0 & 0xffffu) << 16);           // tensor_dim0[15:0] @ bits 63:48
    g1[2] = (int)((td0 >> 16) | ((td1 & 0xffffu) << 16));
    g1[3] = (int)((td1 >> 16) | (32u << 16));       // tile_dim0 = 32 elems
    g1[4] = (int)(td1 & 0xffffu);                   // tile_dim1 = rows; tile_dim2 = 0
    g1[5] = (int)(st0 & 0xffffffffu);               // tensor_dim0_stride[31:0]
    g1[6] = (int)((st0 >> 32) & 0xffffu);           // tensor_dim0_stride[47:32]
    g1[7] = 0;
    i32x4_t z4 = {0, 0, 0, 0};
    i32x8_t z8 = {0, 0, 0, 0, 0, 0, 0, 0};
    __builtin_amdgcn_tensor_load_to_lds(g0, g1, z4, z4, z8, 0);
}
#endif

// ---------------- precision conversion kernels ----------------

__global__ void __launch_bounds__(256) f32_to_bf16_kernel(const float* __restrict__ src,
                                                          __bf16* __restrict__ dst, int n4) {
    int i = blockIdx.x * 256 + threadIdx.x;
    if (i < n4) {
        float4 v = ((const float4*)src)[i];
        dst[4*i+0] = (__bf16)v.x; dst[4*i+1] = (__bf16)v.y;
        dst[4*i+2] = (__bf16)v.z; dst[4*i+3] = (__bf16)v.w;
    }
}

// LDS-tiled transpose+convert: src K x N (f32) -> dst N x K (bf16); coalesced both sides.
__global__ void __launch_bounds__(256) transpose_cvt_kernel(const float* __restrict__ src,
                                                            __bf16* __restrict__ dst,
                                                            int K, int N) {
    __shared__ float tile[32][33];
    const int k0 = blockIdx.x * 32, n0 = blockIdx.y * 32;
    const int c = threadIdx.x & 31, r0 = threadIdx.x >> 5;
    #pragma unroll
    for (int i = 0; i < 4; i++) {
        int r = r0 + i * 8;
        tile[r][c] = src[(size_t)(k0 + r) * N + n0 + c];
    }
    __syncthreads();
    #pragma unroll
    for (int i = 0; i < 4; i++) {
        int r = r0 + i * 8;
        dst[(size_t)(n0 + r) * K + k0 + c] = (__bf16)tile[c][r];
    }
}

// ---------------- tiled WMMA GEMM: C[M,N] = A[M,K] * Bt[N,K]^T ----------------
// 256 threads = 8 waves; 256x128 output tile; each wave owns 32x128 (16 accumulators),
// so each B fragment feeds two WMMAs. Double-buffered LDS; TDM DMA staging when available.

__device__ __forceinline__ void stage_tiles(__bf16* __restrict__ Asb, __bf16* __restrict__ Bsb,
                                            const __bf16* __restrict__ A,
                                            const __bf16* __restrict__ Bt,
                                            int m0, int n0, int k0, int K, int tid) {
    const int row = tid;                      // A: 256 rows x 32, one row per thread
    #pragma unroll
    for (int j = 0; j < 4; j++)
        *(uint4*)&Asb[row*32 + j*8] = *(const uint4*)&A[(size_t)(m0 + row) * K + k0 + j*8];
    const int brow = tid & 127, bh = (tid >> 7) * 16;   // B: 128 rows x 32, 2 threads/row
    #pragma unroll
    for (int j = 0; j < 2; j++)
        *(uint4*)&Bsb[brow*32 + bh + j*8] = *(const uint4*)&Bt[(size_t)(n0 + brow) * K + k0 + bh + j*8];
}

__global__ void __launch_bounds__(256) gemm_bf16_nt_kernel(const __bf16* __restrict__ A,
                                                           const __bf16* __restrict__ Bt,
                                                           float* __restrict__ C,
                                                           int M, int N, int K) {
    __shared__ __align__(16) __bf16 As[2][256 * 32];
    __shared__ __align__(16) __bf16 Bs[2][128 * 32];
    const int tid = threadIdx.x, wave = tid >> 5, lane = tid & 31;
    const int m0 = blockIdx.y * 256, n0 = blockIdx.x * 128;
    const int nsteps = K / 32;

    // prologue: fill buffer 0
#if USE_TDM
    if (wave == 0) {
        tdm_load_tile(A  + (size_t)m0 * K, As[0], 256, K);
        tdm_load_tile(Bt + (size_t)n0 * K, Bs[0], 128, K);
        __builtin_amdgcn_s_wait_tensorcnt(0);
    }
#else
    stage_tiles(As[0], Bs[0], A, Bt, m0, n0, 0, K, tid);
#endif
    __syncthreads();

    v8f acc[2][8] = {};
    for (int step = 0; step < nsteps; ++step) {
        const int p = step & 1;
        // kick off DMA / staging for next tile into the other buffer
        if (step + 1 < nsteps) {
#if USE_TDM
            if (wave == 0) {
                tdm_load_tile(A  + (size_t)m0 * K + (step + 1) * 32, As[p ^ 1], 256, K);
                tdm_load_tile(Bt + (size_t)n0 * K + (step + 1) * 32, Bs[p ^ 1], 128, K);
            }
#else
            stage_tiles(As[p ^ 1], Bs[p ^ 1], A, Bt, m0, n0, (step + 1) * 32, K, tid);
#endif
        }
        // compute on current buffer
        v16bf a0 = ld_frag(&As[p][(wave*32 +      (lane & 15)) * 32], lane);
        v16bf a1 = ld_frag(&As[p][(wave*32 + 16 + (lane & 15)) * 32], lane);
        #pragma unroll
        for (int nt = 0; nt < 8; nt++) {
            v16bf bfrag = ld_frag(&Bs[p][(nt*16 + (lane & 15)) * 32], lane);
            acc[0][nt] = wmma_bf16(a0, bfrag, acc[0][nt]);
            acc[1][nt] = wmma_bf16(a1, bfrag, acc[1][nt]);
        }
#if USE_TDM
        if (wave == 0 && step + 1 < nsteps) __builtin_amdgcn_s_wait_tensorcnt(0);
#endif
        __syncthreads();
    }

    // f32 C/D layout: VGPR r, lane l -> (m = r + 8*(l>>4), n = l&15)
    const int ncol = n0 + (lane & 15);
    #pragma unroll
    for (int half = 0; half < 2; half++) {
        const int mrow = m0 + wave*32 + half*16 + (lane >> 4) * 8;
        #pragma unroll
        for (int nt = 0; nt < 8; nt++)
            #pragma unroll
            for (int rr = 0; rr < 8; rr++)
                C[(size_t)(mrow + rr) * N + ncol + nt*16] = acc[half][nt][rr];
    }
}

// ---------------- RMSNorm + RoPE (+optional scale), f32 -> bf16 head-major ----------------
__global__ void __launch_bounds__(256) rmsnorm_rope_kernel(const float* __restrict__ src,
                                                           const float* __restrict__ w,
                                                           const float* __restrict__ cosT,
                                                           const float* __restrict__ sinT,
                                                           __bf16* __restrict__ dst,
                                                           int nh, float scale) {
    const int tid = threadIdx.x, wave = tid >> 5, lane = tid & 31;
    const int row = blockIdx.x * 8 + wave;        // (b*S+s)*nh + h
    const int h = row % nh, bs = row / nh;
    const int b = bs / S_, s = bs % S_;
    const float* x = src + (size_t)row * D_;
    const int d0 = lane * 4;

    float4 xv = *(const float4*)(x + d0);
    float ss = xv.x*xv.x + xv.y*xv.y + xv.z*xv.z + xv.w*xv.w;
    #pragma unroll
    for (int m = 16; m >= 1; m >>= 1) ss += __shfl_xor(ss, m);
    float rn = rsqrtf(ss * (1.0f / D_) + EPS_);

    float4 w4 = *(const float4*)(w + d0);
    float xn[4] = { xv.x*rn*w4.x, xv.y*rn*w4.y, xv.z*rn*w4.z, xv.w*rn*w4.w };
    float4 c4 = *(const float4*)(cosT + (size_t)s * D_ + d0);
    float4 s4 = *(const float4*)(sinT + (size_t)s * D_ + d0);
    const float cc[4] = { c4.x, c4.y, c4.z, c4.w };
    const float snv[4] = { s4.x, s4.y, s4.z, s4.w };

    __bf16* out = dst + ((size_t)(b * nh + h) * S_ + s) * D_ + d0;
    #pragma unroll
    for (int i = 0; i < 4; i++) {
        float other = __shfl_xor(xn[i], 16);          // partner at d ^ 64
        float rot = (lane < 16) ? -other : other;     // rotate_half
        out[i] = (__bf16)((xn[i] * cc[i] + rot * snv[i]) * scale);
    }
}

// ---------------- V transpose (LDS-tiled): vf (B*S)x(KV*D) f32 -> vT [b,kv,d,s] bf16 -----
__global__ void __launch_bounds__(256) v_transpose_kernel(const float* __restrict__ vf,
                                                          __bf16* __restrict__ vT) {
    __shared__ float tile[32][33];
    const int bk = blockIdx.z;                    // b*KV + kv
    const int b = bk >> 2, kv = bk & 3;
    const int s0 = blockIdx.x * 32, d0 = blockIdx.y * 32;
    const int c = threadIdx.x & 31, r0 = threadIdx.x >> 5;
    #pragma unroll
    for (int i = 0; i < 4; i++) {
        int r = r0 + i * 8;                       // s offset
        tile[r][c] = vf[(size_t)(b * S_ + s0 + r) * (KV_ * D_) + kv * D_ + d0 + c];
    }
    __syncthreads();
    #pragma unroll
    for (int i = 0; i < 4; i++) {
        int r = r0 + i * 8;                       // d offset
        vT[((size_t)bk * D_ + d0 + r) * S_ + s0 + c] = (__bf16)tile[c][r];
    }
}

// ---------------- Flash attention: 128 queries/block, stream K/V in 64-wide tiles --------
__global__ void __launch_bounds__(256) attn_kernel(const __bf16* __restrict__ qb,
                                                   const __bf16* __restrict__ kb,
                                                   const __bf16* __restrict__ vT,
                                                   __bf16* __restrict__ aout) {
    __shared__ __align__(16) __bf16 Ks[64 * 128];   // [t][d]
    __shared__ __align__(16) __bf16 Vs[128 * 64];   // [d][t]  (= V^T tile)
    __shared__ __align__(16) __bf16 Ps[8][16 * 64]; // per-wave P staging
    const int tid = threadIdx.x, wave = tid >> 5, lane = tid & 31;
    const int b = blockIdx.z, h = blockIdx.y, kv = h / G_;
    const int s0 = blockIdx.x * 128;

    const __bf16* qbase = qb + (((size_t)(b * H_ + h)) * S_ + s0 + wave*16 + (lane & 15)) * D_;
    v16bf qf[4];
    #pragma unroll
    for (int kk = 0; kk < 4; kk++) qf[kk] = ld_frag(qbase + kk * 32, lane);

    v8f o[8] = {};
    float rmax[8], rsum[8];
    #pragma unroll
    for (int r = 0; r < 8; r++) { rmax[r] = -1e30f; rsum[r] = 0.f; }

    const __bf16* kbase = kb + (size_t)(b * KV_ + kv) * S_ * D_;
    const __bf16* vbase = vT + (size_t)(b * KV_ + kv) * D_ * S_;

    for (int t0 = 0; t0 < S_; t0 += 64) {
        {
            const int kr = tid >> 2, kcb = (tid & 3) * 32;
            #pragma unroll
            for (int j = 0; j < 4; j++)
                *(uint4*)&Ks[kr*128 + kcb + j*8] =
                    *(const uint4*)&kbase[(size_t)(t0 + kr) * D_ + kcb + j*8];
            const int dr = tid >> 1, tb = (tid & 1) * 32;
            #pragma unroll
            for (int j = 0; j < 4; j++)
                *(uint4*)&Vs[dr*64 + tb + j*8] =
                    *(const uint4*)&vbase[(size_t)dr * S_ + t0 + tb + j*8];
        }
        __syncthreads();

        v8f sc[4];
        #pragma unroll
        for (int nt = 0; nt < 4; nt++) {
            v8f a = {};
            #pragma unroll
            for (int kk = 0; kk < 4; kk++) {
                v16bf bfrag = ld_frag(&Ks[(nt*16 + (lane & 15)) * 128 + kk*32], lane);
                a = wmma_bf16(qf[kk], bfrag, a);
            }
            sc[nt] = a;
        }

        #pragma unroll
        for (int r = 0; r < 8; r++) {
            float tm = sc[0][r];
            #pragma unroll
            for (int nt = 1; nt < 4; nt++) tm = fmaxf(tm, sc[nt][r]);
            #pragma unroll
            for (int m = 8; m >= 1; m >>= 1) tm = fmaxf(tm, __shfl_xor(tm, m));
            float mnew  = fmaxf(rmax[r], tm);
            float alpha = __expf(rmax[r] - mnew);
            float lsum = 0.f;
            #pragma unroll
            for (int nt = 0; nt < 4; nt++) {
                float p = __expf(sc[nt][r] - mnew);
                sc[nt][r] = p;
                lsum += p;
            }
            #pragma unroll
            for (int m = 8; m >= 1; m >>= 1) lsum += __shfl_xor(lsum, m);
            rsum[r] = rsum[r] * alpha + lsum;
            rmax[r] = mnew;
            #pragma unroll
            for (int nt = 0; nt < 8; nt++) o[nt][r] *= alpha;
        }

        {
            const int mr = (lane >> 4) * 8;
            #pragma unroll
            for (int r = 0; r < 8; r++)
                #pragma unroll
                for (int nt = 0; nt < 4; nt++)
                    Ps[wave][(mr + r) * 64 + nt*16 + (lane & 15)] = (__bf16)sc[nt][r];
        }
        asm volatile("s_wait_dscnt 0" ::: "memory");   // wave-local LDS RAW fence

        #pragma unroll
        for (int tt = 0; tt < 2; tt++) {
            v16bf pa = ld_frag(&Ps[wave][(lane & 15) * 64 + tt*32], lane);
            #pragma unroll
            for (int nt = 0; nt < 8; nt++) {
                v16bf vb = ld_frag(&Vs[(nt*16 + (lane & 15)) * 64 + tt*32], lane);
                o[nt] = wmma_bf16(pa, vb, o[nt]);
            }
        }
        __syncthreads();
    }

    const int mbase = s0 + wave*16 + (lane >> 4) * 8;
    #pragma unroll
    for (int r = 0; r < 8; r++) {
        float inv = 1.f / rsum[r];
        int s = mbase + r;
        __bf16* op = aout + (((size_t)(b * S_ + s)) * H_ + h) * D_ + (lane & 15);
        #pragma unroll
        for (int nt = 0; nt < 8; nt++) op[nt*16] = (__bf16)(o[nt][r] * inv);
    }
}

// ---------------- host launch ----------------

extern "C" void kernel_launch(void* const* d_in, const int* in_sizes, int n_in,
                              void* d_out, int out_size, void* d_ws, size_t ws_size,
                              hipStream_t stream) {
    (void)in_sizes; (void)n_in; (void)out_size; (void)ws_size;
    const float* x    = (const float*)d_in[0];
    const float* cosT = (const float*)d_in[1];
    const float* sinT = (const float*)d_in[2];
    const float* Wq   = (const float*)d_in[3];
    const float* Wk   = (const float*)d_in[4];
    const float* Wv   = (const float*)d_in[5];
    const float* Wo   = (const float*)d_in[6];
    const float* qw   = (const float*)d_in[7];
    const float* kw   = (const float*)d_in[8];
    float* out = (float*)d_out;

    const size_t MS = (size_t)B_ * S_;          // 8192 rows
    char* ws = (char*)d_ws;
    auto carve = [&](size_t bytes) -> char* {
        char* p = ws; ws += (bytes + 255) & ~(size_t)255; return p;
    };
    __bf16* xb  = (__bf16*)carve(MS * HID_ * 2);
    __bf16* WqT = (__bf16*)carve((size_t)HID_ * (H_*D_)  * 2);
    __bf16* WkT = (__bf16*)carve((size_t)HID_ * (KV_*D_) * 2);
    __bf16* WvT = (__bf16*)carve((size_t)HID_ * (KV_*D_) * 2);
    __bf16* WoT = (__bf16*)carve((size_t)(H_*D_) * HID_  * 2);
    float*  qf  = (float*) carve(MS * (H_*D_)  * 4);
    float*  kf  = (float*) carve(MS * (KV_*D_) * 4);
    float*  vf  = (float*) carve(MS * (KV_*D_) * 4);
    __bf16* qbv = (__bf16*)carve(MS * (H_*D_)  * 2);
    __bf16* kbv = (__bf16*)carve(MS * (KV_*D_) * 2);
    __bf16* vT  = (__bf16*)carve(MS * (KV_*D_) * 2);
    __bf16* aout = (__bf16*)qf;                 // reuse qf region (qf dead after q-prep)

    // 1) conversions (coalesced, LDS-tiled transposes)
    f32_to_bf16_kernel<<<(MS * HID_ / 4 + 255) / 256, 256, 0, stream>>>(x, xb, (int)(MS * HID_ / 4));
    transpose_cvt_kernel<<<dim3(HID_/32,    (H_*D_)/32),  256, 0, stream>>>(Wq, WqT, HID_,   H_*D_);
    transpose_cvt_kernel<<<dim3(HID_/32,    (KV_*D_)/32), 256, 0, stream>>>(Wk, WkT, HID_,   KV_*D_);
    transpose_cvt_kernel<<<dim3(HID_/32,    (KV_*D_)/32), 256, 0, stream>>>(Wv, WvT, HID_,   KV_*D_);
    transpose_cvt_kernel<<<dim3((H_*D_)/32, HID_/32),     256, 0, stream>>>(Wo, WoT, H_*D_,  HID_);

    // 2) projections (WMMA GEMMs, TDM-staged double-buffered LDS)
    gemm_bf16_nt_kernel<<<dim3((H_*D_)/128,  MS/256), 256, 0, stream>>>(xb, WqT, qf, (int)MS, H_*D_,  HID_);
    gemm_bf16_nt_kernel<<<dim3((KV_*D_)/128, MS/256), 256, 0, stream>>>(xb, WkT, kf, (int)MS, KV_*D_, HID_);
    gemm_bf16_nt_kernel<<<dim3((KV_*D_)/128, MS/256), 256, 0, stream>>>(xb, WvT, vf, (int)MS, KV_*D_, HID_);

    // 3) RMSNorm + RoPE; SCALE folded into q. V transposed for the P@V WMMA operand.
    rmsnorm_rope_kernel<<<(unsigned)(MS * H_  / 8), 256, 0, stream>>>(qf, qw, cosT, sinT, qbv, H_,  SCALE_);
    rmsnorm_rope_kernel<<<(unsigned)(MS * KV_ / 8), 256, 0, stream>>>(kf, kw, cosT, sinT, kbv, KV_, 1.0f);
    v_transpose_kernel<<<dim3(S_/32, D_/32, B_*KV_), 256, 0, stream>>>(vf, vT);

    // 4) flash attention
    attn_kernel<<<dim3(S_/128, H_, B_), 256, 0, stream>>>(qbv, kbv, vT, aout);

    // 5) output projection
    gemm_bf16_nt_kernel<<<dim3(HID_/128, MS/256), 256, 0, stream>>>(aout, WoT, out, (int)MS, HID_, HID_);
}
